// knn_PAConv_with_norm_49572512530930
// MI455X (gfx1250) — matrix operand order repaired and checked by reading, not compiled, not run on hardware
//
#include <hip/hip_runtime.h>
#include <stdint.h>

#define Bsz  4
#define Np   2048
#define Kn   32
#define CIN  64
#define COUT 64
#define Mm   8
#define HID  16
#define Ptot (Bsz*Np*Kn)      /* 262144 positions */
#define KDIM 512              /* M*CIN folded K dimension */
#define EPSV 1e-5f

typedef __attribute__((ext_vector_type(16))) __bf16 v16bf;
typedef __attribute__((ext_vector_type(8)))  float  v8f;

__device__ __forceinline__ unsigned short f2bf(float f) {
  unsigned u = __float_as_uint(f);
  u += 0x7FFFu + ((u >> 16) & 1u);          // round-to-nearest-even
  return (unsigned short)(u >> 16);
}
__device__ __forceinline__ float bf2f(unsigned short h) {
  return __uint_as_float(((unsigned)h) << 16);
}
__device__ __forceinline__ unsigned pk_mul_bf16(unsigned a, unsigned b) {
  unsigned d;
  asm("v_pk_mul_bf16 %0, %1, %2" : "=v"(d) : "v"(a), "v"(b));
  return d;
}

/* ---------------- workspace layout (bytes) ---------------- */
#define OFF_H      ((size_t)0)                     /* P*16 f32  = 16 MB  */
#define OFF_FEATT  ((size_t)16777216)              /* P*64 bf16 = 32 MB  */
#define OFF_SCORES ((size_t)50331648)              /* P*8  f32  =  8 MB  */
#define OFF_WCM    ((size_t)58720256)              /* 64*512 bf16 = 64KB */
#define OFF_PREOUT ((size_t)58785792)              /* P*64 bf16 = 32 MB  */
#define OFF_STATS  ((size_t)92340224)              /* 160 f32            */

/* stats layout: [0..15] sum_h, [16..31] sumsq_h, [32..95] sum_o, [96..159] sumsq_o */

__global__ void k_zero(float* __restrict__ stats) {
  if (threadIdx.x < 160) stats[threadIdx.x] = 0.0f;
}

/* ---- geometry features + scorenet conv1 + BN stat accumulation ---- */
__global__ void __launch_bounds__(256)
k_geom(const float* __restrict__ gxyz, const float* __restrict__ gdist,
       const float* __restrict__ gnorm, const float* __restrict__ gang,
       const float* __restrict__ w1, float* __restrict__ hbuf,
       float* __restrict__ stats) {
  __shared__ float bsum[HID], bsq[HID];
  if (threadIdx.x < HID) { bsum[threadIdx.x] = 0.0f; bsq[threadIdx.x] = 0.0f; }
  __syncthreads();

  const int pos   = blockIdx.x * 256 + threadIdx.x;
  const int b     = pos >> 16;               /* N*Kn = 65536 */
  const int nk    = pos & 65535;
  const int base3 = b * 196608 + nk;         /* + ch*65536   */
  const int basec = base3 - (nk & 31);       /* k = 0 (center) */

  const float x0 = gxyz[base3], x1 = gxyz[base3 + 65536], x2 = gxyz[base3 + 131072];
  const float n0 = gnorm[base3], n1 = gnorm[base3 + 65536], n2 = gnorm[base3 + 131072];
  const float c0 = gnorm[basec], c1 = gnorm[basec + 65536], c2 = gnorm[basec + 131072];
  const float a0 = gang[base3],  a1 = gang[base3 + 65536],  a2 = gang[base3 + 131072];
  const float d0 = n0 - c0, d1 = n1 - c1, d2 = n2 - c2;
  const float inv = 1.0f / (1.0f + gdist[pos]);

  float f[14];
  f[0]  = x0 * inv;  f[1] = x1 * inv;  f[2] = x2 * inv;
  f[3]  = sqrtf(x0*x0 + x1*x1 + x2*x2) * inv;
  f[4]  = n0 * inv;  f[5] = n1 * inv;  f[6] = n2 * inv;
  f[7]  = d0 * inv;  f[8] = d1 * inv;  f[9] = d2 * inv;
  f[10] = sqrtf(d0*d0 + d1*d1 + d2*d2) * inv;
  f[11] = a0 * inv;  f[12] = a1 * inv;  f[13] = a2 * inv;

  float h[16];
#pragma unroll
  for (int o = 0; o < 16; ++o) {
    float acc = 0.0f;
#pragma unroll
    for (int c = 0; c < 14; ++c) acc += w1[o * 14 + c] * f[c];
    h[o] = acc;
  }
  float4* hp = (float4*)(hbuf + (size_t)pos * 16);
  hp[0] = make_float4(h[0],  h[1],  h[2],  h[3]);
  hp[1] = make_float4(h[4],  h[5],  h[6],  h[7]);
  hp[2] = make_float4(h[8],  h[9],  h[10], h[11]);
  hp[3] = make_float4(h[12], h[13], h[14], h[15]);

#pragma unroll
  for (int o = 0; o < 16; ++o) {
    float r = h[o], rq = h[o] * h[o];
#pragma unroll
    for (int off = 16; off >= 1; off >>= 1) {
      r  += __shfl_xor(r,  off);
      rq += __shfl_xor(rq, off);
    }
    if ((threadIdx.x & 31) == 0) { atomicAdd(&bsum[o], r); atomicAdd(&bsq[o], rq); }
  }
  __syncthreads();
  if (threadIdx.x < HID) {
    atomicAdd(&stats[threadIdx.x],      bsum[threadIdx.x]);
    atomicAdd(&stats[16 + threadIdx.x], bsq[threadIdx.x]);
  }
}

/* ---- transpose feature (B,CIN,N,K) -> position-major (P, CIN) bf16 ---- */
__global__ void __launch_bounds__(256)
k_tfeat(const float* __restrict__ feat, unsigned short* __restrict__ featT) {
  const int bn = blockIdx.x, b = bn >> 11, n = bn & 2047;
  __shared__ unsigned short tile[64][33];
  const float* src = feat + (size_t)b * 4194304 + (size_t)n * 32;
#pragma unroll
  for (int i = 0; i < 8; ++i) {
    int lin = threadIdx.x + i * 256;
    int c = lin >> 5, k = lin & 31;
    tile[c][k] = f2bf(src[(size_t)c * 65536 + k]);
  }
  __syncthreads();
  unsigned short* dst = featT + (size_t)bn * 32 * 64;
#pragma unroll
  for (int i = 0; i < 8; ++i) {
    int lin = threadIdx.x + i * 256;
    int k = lin >> 6, c = lin & 63;
    dst[lin] = tile[c][k];
  }
}

/* ---- weightbank (CIN, M*COUT) -> W' column-major bf16: Wcm[n*512 + m*64 + c] ---- */
__global__ void k_wswz(const float* __restrict__ wb, unsigned short* __restrict__ Wcm) {
  const int tid = blockIdx.x * 256 + threadIdx.x;   /* 32768 */
  const int nn = tid >> 9, j = tid & 511;
  const int m = j >> 6, c = j & 63;
  Wcm[tid] = f2bf(wb[c * 512 + m * 64 + nn]);
}

/* ---- BN(h)+ReLU, conv2, softmax_one -> scores ---- */
__global__ void __launch_bounds__(256)
k_score(const float* __restrict__ hbuf, const float* __restrict__ stats,
        const float* __restrict__ g1, const float* __restrict__ b1,
        const float* __restrict__ w2, const float* __restrict__ bias2,
        float* __restrict__ scores) {
  const int pos = blockIdx.x * 256 + threadIdx.x;
  const float invP = 1.0f / (float)Ptot;
  const float4* hp = (const float4*)(hbuf + (size_t)pos * 16);
  float4 hv[4] = {hp[0], hp[1], hp[2], hp[3]};
  const float* hvf = (const float*)hv;
  float hn[16];
#pragma unroll
  for (int o = 0; o < 16; ++o) {
    float mean = stats[o] * invP;
    float var  = stats[16 + o] * invP - mean * mean;
    float sc   = g1[o] * rsqrtf(var + EPSV);
    float sh   = b1[o] - mean * sc;
    hn[o] = fmaxf(0.0f, hvf[o] * sc + sh);
  }
  float s[8];
#pragma unroll
  for (int m = 0; m < 8; ++m) {
    float acc = bias2[m];
#pragma unroll
    for (int o = 0; o < 16; ++o) acc += w2[m * 16 + o] * hn[o];
    s[m] = acc;
  }
  float mx = s[0];
#pragma unroll
  for (int m = 1; m < 8; ++m) mx = fmaxf(mx, s[m]);
  float den = __expf(-mx);
  float e[8];
#pragma unroll
  for (int m = 0; m < 8; ++m) { e[m] = __expf(s[m] - mx); den += e[m]; }
  const float r = 1.0f / den;
  float4* sp = (float4*)(scores + (size_t)pos * 8);
  sp[0] = make_float4(e[0] * r, e[1] * r, e[2] * r, e[3] * r);
  sp[1] = make_float4(e[4] * r, e[5] * r, e[6] * r, e[7] * r);
}

/* ---- main WMMA GEMM: out(P x 64) = A(P x 512) x W'(512 x 64), A built on the fly ---- */
__global__ void __launch_bounds__(256)
k_gemm(const unsigned short* __restrict__ featT, const float* __restrict__ scores,
       const unsigned short* __restrict__ Wcm, unsigned short* __restrict__ preout,
       float* __restrict__ statsO) {
  __shared__ unsigned short Wlds[COUT * KDIM];   /* 64 KB, column-major W' */
  __shared__ float bsum[COUT], bsq[COUT];
  {
    const uint4* src = (const uint4*)Wcm;
    uint4* dst = (uint4*)Wlds;
#pragma unroll
    for (int i = 0; i < 16; ++i) dst[threadIdx.x + i * 256] = src[threadIdx.x + i * 256];
  }
  if (threadIdx.x < COUT) { bsum[threadIdx.x] = 0.0f; bsq[threadIdx.x] = 0.0f; }
  __syncthreads();

  const int w = threadIdx.x >> 5, lane = threadIdx.x & 31;
  const int half = lane >> 4, ln = lane & 15;
  const int rowbase = blockIdx.x * 256 + w * 32;  /* 32 rows per wave */

  /* per-lane feature rows: only the 16 bf16-pairs this lane-half needs,
     pre-selected so all register indexing below is compile-time */
  unsigned frS[2][16];
  unsigned sc[2][8];
#pragma unroll
  for (int g = 0; g < 2; ++g) {
    const int r = rowbase + g * 16 + ln;
    const uint4* fp = (const uint4*)(featT + (size_t)r * 64);
#pragma unroll
    for (int i = 0; i < 4; ++i) {
      uint4 q = fp[2 * i + half];
      frS[g][i * 4 + 0] = q.x; frS[g][i * 4 + 1] = q.y;
      frS[g][i * 4 + 2] = q.z; frS[g][i * 4 + 3] = q.w;
    }
    const float4* spn = (const float4*)(scores + (size_t)r * 8);
    float4 s0 = spn[0], s1 = spn[1];
    const float ss[8] = {s0.x, s0.y, s0.z, s0.w, s1.x, s1.y, s1.z, s1.w};
#pragma unroll
    for (int m = 0; m < 8; ++m) {
      unsigned hb = (unsigned)f2bf(ss[m]);
      sc[g][m] = (hb << 16) | hb;                 /* packed (s,s) bf16 pair */
    }
  }

  v8f acc[2][4];
#pragma unroll
  for (int g = 0; g < 2; ++g)
#pragma unroll
    for (int nt = 0; nt < 4; ++nt) { v8f z = {}; acc[g][nt] = z; }

#pragma unroll
  for (int kk = 0; kk < 16; ++kk) {               /* K = 512, step 32 */
    const int m  = kk >> 1;                       /* score index for this chunk */
    const int q0 = (kk & 1) * 8;                  /* frS offset for c0 = 0 / 32 */
    union { unsigned u[8]; v16bf v; } a[2];
#pragma unroll
    for (int g = 0; g < 2; ++g) {
#pragma unroll
      for (int i = 0; i < 4; ++i) {
        a[g].u[i]     = pk_mul_bf16(sc[g][m], frS[g][q0 + i]);
        a[g].u[4 + i] = pk_mul_bf16(sc[g][m], frS[g][q0 + 4 + i]);
      }
    }
#pragma unroll
    for (int nt = 0; nt < 4; ++nt) {
      union { uint4 q[2]; v16bf v; } bf;
      const uint4* bp = (const uint4*)(Wlds + (nt * 16 + ln) * KDIM + kk * 32);
      bf.q[0] = bp[half]; bf.q[1] = bp[2 + half]; /* 2x ds_load_b128 per B fragment */
      acc[0][nt] = __builtin_amdgcn_wmma_f32_16x16x32_bf16(
          false, a[0].v, false, bf.v, (short)0, acc[0][nt], false, false);
      acc[1][nt] = __builtin_amdgcn_wmma_f32_16x16x32_bf16(
          false, a[1].v, false, bf.v, (short)0, acc[1][nt], false, false);
    }
  }

  /* store pre-BN output (bf16) + per-column stats for output BN */
#pragma unroll
  for (int nt = 0; nt < 4; ++nt) {
    float cs = 0.0f, cq = 0.0f;
    const int col = nt * 16 + ln;
#pragma unroll
    for (int g = 0; g < 2; ++g) {
      const int rb = rowbase + g * 16 + half * 8;
#pragma unroll
      for (int v = 0; v < 8; ++v) {
        float x = acc[g][nt][v];
        preout[(size_t)(rb + v) * 64 + col] = f2bf(x);
        cs += x; cq += x * x;
      }
    }
    cs += __shfl_xor(cs, 16);                     /* lane l and l+16 share N=col */
    cq += __shfl_xor(cq, 16);
    if (half == 0) { atomicAdd(&bsum[col], cs); atomicAdd(&bsq[col], cq); }
  }
  __syncthreads();
  if (threadIdx.x < COUT) {
    atomicAdd(&statsO[threadIdx.x],        bsum[threadIdx.x]);
    atomicAdd(&statsO[COUT + threadIdx.x], bsq[threadIdx.x]);
  }
}

/* ---- final BN + ReLU + transpose back to (B, COUT, N, K) ---- */
__global__ void __launch_bounds__(256)
k_bnout(const unsigned short* __restrict__ preout, const float* __restrict__ statsO,
        const float* __restrict__ gamma, const float* __restrict__ beta,
        float* __restrict__ out) {
  const int bn = blockIdx.x, b = bn >> 11, n = bn & 2047;
  __shared__ float scale[64], shift[64];
  __shared__ float tile[64][33];
  const float invP = 1.0f / (float)Ptot;
  if (threadIdx.x < 64) {
    const int c = threadIdx.x;
    float mean = statsO[c] * invP;
    float var  = statsO[64 + c] * invP - mean * mean;
    float scl  = gamma[c] * rsqrtf(var + EPSV);
    scale[c] = scl; shift[c] = beta[c] - mean * scl;
  }
  __syncthreads();
  const unsigned short* src = preout + (size_t)bn * 32 * 64;
#pragma unroll
  for (int i = 0; i < 8; ++i) {
    int lin = threadIdx.x + i * 256;
    int k = lin >> 6, c = lin & 63;
    float v = bf2f(src[lin]);
    tile[c][k] = fmaxf(0.0f, v * scale[c] + shift[c]);
  }
  __syncthreads();
  float* dst = out + (size_t)b * 4194304 + (size_t)n * 32;
#pragma unroll
  for (int i = 0; i < 8; ++i) {
    int lin = threadIdx.x + i * 256;
    int c = lin >> 5, k = lin & 31;
    dst[(size_t)c * 65536 + k] = tile[c][k];
  }
}

extern "C" void kernel_launch(void* const* d_in, const int* in_sizes, int n_in,
                              void* d_out, int out_size, void* d_ws, size_t ws_size,
                              hipStream_t stream) {
  const float* gxyz  = (const float*)d_in[0];
  const float* gdist = (const float*)d_in[1];
  const float* gnorm = (const float*)d_in[2];
  const float* gang  = (const float*)d_in[3];
  const float* feat  = (const float*)d_in[4];
  const float* w1    = (const float*)d_in[5];
  const float* g1    = (const float*)d_in[6];
  const float* b1    = (const float*)d_in[7];
  const float* w2    = (const float*)d_in[8];
  const float* bias2 = (const float*)d_in[9];
  const float* wb    = (const float*)d_in[10];
  const float* gamma = (const float*)d_in[11];
  const float* beta  = (const float*)d_in[12];

  char* ws = (char*)d_ws;
  float*          hbuf   = (float*)(ws + OFF_H);
  unsigned short* featT  = (unsigned short*)(ws + OFF_FEATT);
  float*          scores = (float*)(ws + OFF_SCORES);
  unsigned short* Wcm    = (unsigned short*)(ws + OFF_WCM);
  unsigned short* preout = (unsigned short*)(ws + OFF_PREOUT);
  float*          stats  = (float*)(ws + OFF_STATS);

  k_zero<<<1, 256, 0, stream>>>(stats);
  k_geom<<<Ptot / 256, 256, 0, stream>>>(gxyz, gdist, gnorm, gang, w1, hbuf, stats);
  k_tfeat<<<Bsz * Np, 256, 0, stream>>>(feat, featT);
  k_wswz<<<128, 256, 0, stream>>>(wb, Wcm);
  k_score<<<Ptot / 256, 256, 0, stream>>>(hbuf, stats, g1, b1, w2, bias2, scores);
  k_gemm<<<Ptot / 256, 256, 0, stream>>>(featT, scores, Wcm, preout, stats + 32);
  k_bnout<<<Bsz * Np, 256, 0, stream>>>(preout, stats + 32, gamma, beta, (float*)d_out);
}
